// HCSASelfAttention_53635551592656
// MI455X (gfx1250) — matrix operand order repaired and checked by reading, not compiled, not run on hardware
//
#include <hip/hip_runtime.h>

#define N_EMBD   1024
#define N_HEADS  16
#define HEAD_DIM 64
#define T_SEQ    1024
#define LDS_PITCH 80   // bytes per 32-element bf16 row after TDM padding (64B data + 16B pad)

typedef __attribute__((ext_vector_type(16))) __bf16 v16bf;
typedef __attribute__((ext_vector_type(8)))  __bf16 v8bf;
typedef __attribute__((ext_vector_type(2)))  __bf16 v2bf;
typedef __attribute__((ext_vector_type(8)))  float  v8f;
typedef __attribute__((ext_vector_type(4)))  unsigned int v4u;
typedef __attribute__((ext_vector_type(8)))  int    v8i;
typedef __attribute__((ext_vector_type(4)))  int    v4i;

// ---------------------------------------------------------------------------
// fp32 -> bf16 conversion (row-major copy)
// ---------------------------------------------------------------------------
__global__ void cvt_bf16_kernel(const float* __restrict__ in,
                                __bf16* __restrict__ out, int n) {
  int i = blockIdx.x * blockDim.x + threadIdx.x;
  if (i < n) out[i] = (__bf16)in[i];
}

// fp32 [rows,cols] -> bf16 transposed [cols,rows] (so WMMA B fragments are
// contiguous K-runs per output-feature row)
__global__ void cvtT_bf16_kernel(const float* __restrict__ in,
                                 __bf16* __restrict__ out, int rows, int cols) {
  int i = blockIdx.x * blockDim.x + threadIdx.x;
  if (i >= rows * cols) return;
  int r = i / cols, c = i - r * cols;
  out[(size_t)c * rows + r] = (__bf16)in[i];
}

// ---------------------------------------------------------------------------
// Tensor Data Mover: DMA a 64-row x 32-col bf16 tile of A (row-major, leading
// dim K) into LDS at byte offset lds_off, with 16B padding per 64B row so the
// LDS pitch is 80B (bank-conflict-free ds_load_b128 fragment reads).
// D# layout per CDNA5 ISA ch.8 (2D tensor, groups 2/3 zero).
// ---------------------------------------------------------------------------
__device__ __forceinline__ void tdm_load_a_tile(const __bf16* gsrc,
                                                unsigned int lds_off,
                                                int K, int M) {
  unsigned long long ga = (unsigned long long)(uintptr_t)gsrc;
  v4u g0;
  g0[0] = 1u;                                   // count=1, no gather, user mode
  g0[1] = lds_off;                              // lds_addr (bytes)
  g0[2] = (unsigned int)ga;                     // global_addr[31:0]
  g0[3] = ((unsigned int)(ga >> 32) & 0x01FFFFFFu) | (2u << 30); // addr[56:32] | type=2
  v8i g1;
  g1[0] = (int)((1u << 16)                      // data_size = 2 bytes
              | (1u << 20)                      // pad_enable
              | (3u << 22)                      // pad_interval: 16 DWORDs (one 64B row)
              | (3u << 25));                    // pad_amount: 4 DWORDs (16B)
  g1[1] = (int)(((unsigned int)K & 0xFFFFu) << 16);        // tensor_dim0[15:0] @ bits 63:48
  g1[2] = (int)((((unsigned int)K >> 16) & 0xFFFFu)        // tensor_dim0[31:16]
              | (((unsigned int)M & 0xFFFFu) << 16));      // tensor_dim1[15:0]
  g1[3] = (int)((((unsigned int)M >> 16) & 0xFFFFu)        // tensor_dim1[31:16]
              | (32u << 16));                              // tile_dim0 = 32 elems
  g1[4] = (int)64u;                                        // tile_dim1 = 64, tile_dim2 = 0
  g1[5] = (int)K;                                          // tensor_dim0_stride[31:0]
  g1[6] = 0;                                               // stride hi | dim1_stride lo
  g1[7] = 0;                                               // dim1_stride hi
  v4i z4 = {0, 0, 0, 0};
#if defined(__clang_major__) && (__clang_major__ >= 23)
  v8i z8 = {0, 0, 0, 0, 0, 0, 0, 0};
  __builtin_amdgcn_tensor_load_to_lds(g0, g1, z4, z4, z8, 0);
#else
  __builtin_amdgcn_tensor_load_to_lds(g0, g1, z4, z4, 0);
#endif
}

// ---------------------------------------------------------------------------
// C[M,N] (fp32) = A[M,K] (bf16, row-major) * Bt[N,K]^T (bf16, row-major)
// One wave computes a 64x64 tile: 4x4 fragments of 16x16, K-step 32.
// Block = 128 threads = 4 waves spread along N; all 4 waves share one A tile,
// staged into LDS by the TDM (double-buffered, DMA overlapped with WMMA).
// B tiles are wave-private and stream through L2 via global_load_b128.
// ---------------------------------------------------------------------------
__global__ void __launch_bounds__(128)
gemm_bf16_wmma_kernel(const __bf16* __restrict__ A, const __bf16* __restrict__ Bt,
                      float* __restrict__ C, int M, int N, int K) {
  __shared__ __align__(16) char aslab[2][64 * LDS_PITCH];

  const int lane  = threadIdx.x & 31;
  const int wave  = threadIdx.x >> 5;
  const int tileN = (blockIdx.x * 4 + wave) * 64;
  const int tileM = blockIdx.y * 64;
  const int lh    = lane & 15;
  const bool hiHalf = lane >= 16;

  v8f acc[4][4] = {};
  const int aShift = hiHalf ? 8 : 0;   // A lane-half K base (ISA 16-bit A layout)
  const int bShift = hiHalf ? 8 : 0;   // B lane-half K base

  if (threadIdx.x == 0)
    tdm_load_a_tile(A + (size_t)tileM * K,
                    (unsigned int)(uintptr_t)&aslab[0][0], K, M);

  for (int k0 = 0; k0 < K; k0 += 32) {
    const int cur = (k0 >> 5) & 1;
    if (threadIdx.x == 0) __builtin_amdgcn_s_wait_tensorcnt(0);
    __syncthreads();                                   // aslab[cur] ready for all waves
    if (threadIdx.x == 0 && (k0 + 32) < K)             // prefetch next slab via TDM
      tdm_load_a_tile(A + (size_t)tileM * K + (k0 + 32),
                      (unsigned int)(uintptr_t)&aslab[cur ^ 1][0], K, M);

    v16bf afrag[4], bfrag[4];
#pragma unroll
    for (int mf = 0; mf < 4; ++mf) {
      const char* ap = &aslab[cur][(mf * 16 + lh) * LDS_PITCH + aShift * 2];
      v8bf lo = *(const v8bf*)ap;            // K = base+0..7
      v8bf hi = *(const v8bf*)(ap + 32);     // K = base+16..23
#pragma unroll
      for (int e = 0; e < 8; ++e) { afrag[mf][e] = lo[e]; afrag[mf][8 + e] = hi[e]; }
    }
#pragma unroll
    for (int nf = 0; nf < 4; ++nf) {
      const __bf16* bp = Bt + (size_t)(tileN + nf * 16 + lh) * K + k0 + bShift;
      v8bf lo = *(const v8bf*)bp;            // K = base+0..7
      v8bf hi = *(const v8bf*)(bp + 16);     // K = base+16..23
#pragma unroll
      for (int e = 0; e < 8; ++e) { bfrag[nf][e] = lo[e]; bfrag[nf][8 + e] = hi[e]; }
    }
#pragma unroll
    for (int mf = 0; mf < 4; ++mf)
#pragma unroll
      for (int nf = 0; nf < 4; ++nf)
        acc[mf][nf] = __builtin_amdgcn_wmma_f32_16x16x32_bf16(
            false, afrag[mf], false, bfrag[nf], (short)0, acc[mf][nf], false, false);

    __syncthreads();                                   // all waves done with aslab[cur]
  }

  const int rowBase = hiHalf ? 8 : 0;                  // C/D layout: VGPR i -> row i (+8)
#pragma unroll
  for (int mf = 0; mf < 4; ++mf)
#pragma unroll
    for (int nf = 0; nf < 4; ++nf) {
      const int col = tileN + nf * 16 + lh;
#pragma unroll
      for (int r = 0; r < 8; ++r)
        C[(size_t)(tileM + mf * 16 + rowBase + r) * N + col] = acc[mf][nf][r];
    }
}

// ---------------------------------------------------------------------------
// Gathered sparse attention: one wave32 per (head, token).
// qkv fp32 [T, 3*C]; each lane owns dims {2*lane, 2*lane+1} of the head.
// Online softmax over the (uniform-per-wave) neighbor list; cross-lane dot
// reduction via shfl_xor. Output y written directly as bf16 for GEMM2.
// ---------------------------------------------------------------------------
__global__ void __launch_bounds__(256)
attn_gather_kernel(const float* __restrict__ qkv, const int* __restrict__ nidx,
                   __bf16* __restrict__ yb, int D) {
  const int lane = threadIdx.x & 31;
  const int wid  = (blockIdx.x * blockDim.x + threadIdx.x) >> 5;
  const int t = wid & (T_SEQ - 1);
  const int h = wid >> 10;

  const float* qrow = qkv + (size_t)t * 3 * N_EMBD + h * HEAD_DIM;
  const float2 qv = *(const float2*)(qrow + 2 * lane);

  float m = -__builtin_inff();
  float l = 0.f, accx = 0.f, accy = 0.f;
  const int* row = nidx + ((size_t)h * T_SEQ + t) * D;

  for (int n = 0; n < D; ++n) {
    const int idx = row[n];                   // uniform across the wave
    if (idx < 0 || idx > t) continue;         // invalid or non-causal
    const float* krow = qkv + (size_t)idx * 3 * N_EMBD + N_EMBD + h * HEAD_DIM;
    const float2 kv = *(const float2*)(krow + 2 * lane);
    float p = qv.x * kv.x + qv.y * kv.y;
#pragma unroll
    for (int off = 16; off >= 1; off >>= 1) p += __shfl_xor(p, off, 32);
    const float s = p * 0.125f;               // 1/sqrt(64)
    const float mnew = fmaxf(m, s);
    const float c = __expf(m - mnew);         // exp(-inf)=0 on first hit
    const float w = __expf(s - mnew);
    const float* vrow = qkv + (size_t)idx * 3 * N_EMBD + 2 * N_EMBD + h * HEAD_DIM;
    const float2 vv = *(const float2*)(vrow + 2 * lane);
    l    = l    * c + w;
    accx = accx * c + w * vv.x;
    accy = accy * c + w * vv.y;
    m = mnew;
  }

  const float inv = 1.0f / l;                 // self is always valid -> l > 0
  v2bf o;
  o.x = (__bf16)(accx * inv);
  o.y = (__bf16)(accy * inv);
  *(v2bf*)(yb + (size_t)t * N_EMBD + h * HEAD_DIM + 2 * lane) = o;
}

// ---------------------------------------------------------------------------
// Launch: convert -> GEMM1(qkv) -> attention -> GEMM2(out)
// ---------------------------------------------------------------------------
extern "C" void kernel_launch(void* const* d_in, const int* in_sizes, int n_in,
                              void* d_out, int out_size, void* d_ws, size_t ws_size,
                              hipStream_t stream) {
  const float* x    = (const float*)d_in[0];   // [T, C]
  const float* Wqkv = (const float*)d_in[1];   // [C, 3C]
  const float* Wout = (const float*)d_in[2];   // [C, C]
  const int*   nidx = (const int*)d_in[3];     // [H, T, D]
  const int D = in_sizes[3] / (N_HEADS * T_SEQ);

  char* ws = (char*)d_ws;
  __bf16* xb    = (__bf16*)(ws);                               //  2 MB
  __bf16* WqkvT = (__bf16*)(ws + (size_t) 2 * 1024 * 1024);    //  6 MB [3C, C]
  __bf16* WoutT = (__bf16*)(ws + (size_t) 8 * 1024 * 1024);    //  2 MB [C, C]
  float*  qkv   = (float*) (ws + (size_t)10 * 1024 * 1024);    // 12 MB [T, 3C]
  __bf16* yb    = (__bf16*)(ws + (size_t)22 * 1024 * 1024);    //  2 MB [T, C]
  float*  out   = (float*)d_out;                               // [T, C] fp32

  cvt_bf16_kernel<<<(T_SEQ * N_EMBD) / 256, 256, 0, stream>>>(x, xb, T_SEQ * N_EMBD);
  cvtT_bf16_kernel<<<(N_EMBD * 3 * N_EMBD) / 256, 256, 0, stream>>>(Wqkv, WqkvT, N_EMBD, 3 * N_EMBD);
  cvtT_bf16_kernel<<<(N_EMBD * N_EMBD) / 256, 256, 0, stream>>>(Wout, WoutT, N_EMBD, N_EMBD);

  dim3 g1((3 * N_EMBD) / 256, T_SEQ / 64);
  gemm_bf16_wmma_kernel<<<g1, 128, 0, stream>>>(xb, WqkvT, qkv, T_SEQ, 3 * N_EMBD, N_EMBD);

  attn_gather_kernel<<<(N_HEADS * T_SEQ) / 8, 256, 0, stream>>>(qkv, nidx, yb, D);

  dim3 g2(N_EMBD / 256, T_SEQ / 64);
  gemm_bf16_wmma_kernel<<<g2, 128, 0, stream>>>(yb, WoutT, out, T_SEQ, N_EMBD, N_EMBD);
}
